// AttnDecoderRNN_27487790695242
// MI455X (gfx1250) — matrix operand order repaired and checked by reading, not compiled, not run on hardware
//
#include <hip/hip_runtime.h>

// ---------------- problem constants ----------------
#define BB   128
#define VV   30000
#define EE   512
#define DD   512
#define AAU  512          // attention dim
#define ENCC 2048
#define HWP  196          // 14*14
#define NBOXX 20
#define PP   216          // 196 + 20
#define TT   20
#define TCAPP 21
#define XHK  3072         // E + ENC + D  (fused LSTM input)

typedef __bf16 bf16;
typedef __attribute__((ext_vector_type(16))) __bf16 v16bf;
typedef __attribute__((ext_vector_type(8)))  __bf16 v8bf;
typedef __attribute__((ext_vector_type(8)))  float  v8f;

__device__ __forceinline__ float sigf(float x) { return 1.f / (1.f + __expf(-x)); }

// ================= register-blocked bf16 WMMA GEMM =================
// C[M,N] = act( A[M,K](bf16,row-major) * Bt[N,K](bf16, pre-transposed) + bias ) * rowmask
// Each wave computes an MT x NT grid of 16x16 tiles; A fragments are reused
// across NT columns and B fragments across MT rows -> (MT*NT)/(MT+NT) WMMAs
// per 32B fragment load instead of 1/2.
template <int MT, int NT>
__global__ __launch_bounds__(256)
void gemm_bf16_blk(const bf16* __restrict__ A, int lda,
                   const bf16* __restrict__ Bt, int ldb,
                   void* __restrict__ Cv, long long ldc, int c_is_bf16,
                   const float* __restrict__ bias,
                   int M, int N, int K, int act,
                   const int* __restrict__ lens, int tcur)
{
    const int nbM = (M >> 4) / MT;
    const int nbN = (N >> 4) / NT;
    const int nBlk = nbM * nbN;
    const int wave = (blockIdx.x << 3) + (threadIdx.x >> 5);
    if (wave >= nBlk) return;                 // wave-uniform exit (EXEC all-ones inside)
    // adjacent waves share bn -> B fragments hit L1 within a WGP
    const int bn = wave / nbM;
    const int bm = wave - bn * nbM;
    const int lane = threadIdx.x & 31;
    const int half = lane >> 4;
    const int l15  = lane & 15;

    const bf16* __restrict__ Arow[MT];
    const bf16* __restrict__ Bcol[NT];
#pragma unroll
    for (int mi = 0; mi < MT; ++mi)
        Arow[mi] = A + (size_t)((bm * MT + mi) * 16 + l15) * lda;
#pragma unroll
    for (int ni = 0; ni < NT; ++ni)
        Bcol[ni] = Bt + (size_t)((bn * NT + ni) * 16 + l15) * ldb + half * 16;

    v8f acc[MT][NT] = {};
    for (int k0 = 0; k0 < K; k0 += 32) {
        v16bf af[MT];
        v16bf bfr[NT];
#pragma unroll
        for (int mi = 0; mi < MT; ++mi) {
            // A frag: lane half h -> K in {k0+half*16..+15} (contiguous)
            v8bf alo = *(const v8bf*)(Arow[mi] + k0 + half * 16);
            v8bf ahi = *(const v8bf*)(Arow[mi] + k0 + half * 16 + 8);
            af[mi] = __builtin_shufflevector(alo, ahi,
                        0,1,2,3,4,5,6,7,8,9,10,11,12,13,14,15);
        }
#pragma unroll
        for (int ni = 0; ni < NT; ++ni) {
            bfr[ni] = *(const v16bf*)(Bcol[ni] + k0);                  // 16 contiguous K
            __builtin_prefetch((const void*)(Bcol[ni] + k0 + 256), 0, 1); // global_prefetch_b8
        }
#pragma unroll
        for (int mi = 0; mi < MT; ++mi)
#pragma unroll
            for (int ni = 0; ni < NT; ++ni)
                acc[mi][ni] = __builtin_amdgcn_wmma_f32_16x16x32_bf16(
                                  false, af[mi], false, bfr[ni],
                                  (short)0, acc[mi][ni], false, false);
    }

#pragma unroll
    for (int ni = 0; ni < NT; ++ni) {
        const int col = (bn * NT + ni) * 16 + l15;
        const float bv = bias ? bias[col] : 0.f;
#pragma unroll
        for (int mi = 0; mi < MT; ++mi) {
#pragma unroll
            for (int r = 0; r < 8; ++r) {
                const int row = (bm * MT + mi) * 16 + half * 8 + r;   // C: VGPR r -> M=half*8+r
                float v = acc[mi][ni][r] + bv;
                if (act == 1) v = sigf(v);
                if (lens) v *= (tcur < lens[row] - 1) ? 1.f : 0.f;
                if (c_is_bf16) ((bf16*)Cv)[(long long)row * ldc + col] = (bf16)v;
                else           ((float*)Cv)[(long long)row * ldc + col] = v;
            }
        }
    }
}

// ================= prep kernels =================
__global__ __launch_bounds__(256)
void build_enc(const float* __restrict__ eo, const float* __restrict__ box,
               bf16* __restrict__ enc, long long total)
{
    long long i = (long long)blockIdx.x * 256 + threadIdx.x;
    if (i >= total) return;
    int c = (int)(i % ENCC);
    long long r = i / ENCC;
    int p = (int)(r % PP);
    int b = (int)(r / PP);
    float v = (p < HWP)
        ? eo [((size_t)b * ENCC + c) * HWP   + p]
        : box[((size_t)b * ENCC + c) * NBOXX + (p - HWP)];
    enc[i] = (bf16)v;
}

// transpose+convert: dst[n*ldd + koff + k] = (bf16) src[k*N + n]
__global__ __launch_bounds__(256)
void tconv(const float* __restrict__ src, bf16* __restrict__ dst,
           int K, int N, int ldd, int koff)
{
    long long i = (long long)blockIdx.x * 256 + threadIdx.x;
    if (i >= (long long)K * N) return;
    int k = (int)(i % K);
    int n = (int)(i / K);
    dst[(size_t)n * ldd + koff + k] = (bf16)src[(size_t)k * N + n];
}

__global__ __launch_bounds__(256)
void mean_kernel(const bf16* __restrict__ enc, bf16* __restrict__ meanb)
{
    int b = blockIdx.y;
    int c = blockIdx.x * 256 + threadIdx.x;
    const bf16* base = enc + (size_t)b * PP * ENCC + c;
    float s = 0.f;
    for (int p = 0; p < PP; ++p) s += (float)base[(size_t)p * ENCC];
    meanb[(size_t)b * ENCC + c] = (bf16)(s * (1.f / (float)PP));
}

__global__ __launch_bounds__(256)
void f2b(const float* __restrict__ s, bf16* __restrict__ d, int n)
{
    int i = blockIdx.x * 256 + threadIdx.x;
    if (i < n) d[i] = (bf16)s[i];
}

__global__ __launch_bounds__(256)
void vadd(const float* __restrict__ a, const float* __restrict__ b,
          float* __restrict__ d, int n)
{
    int i = blockIdx.x * 256 + threadIdx.x;
    if (i < n) d[i] = a[i] + b[i];
}

// ================= per-step kernels =================
// e[b,p] = sum_a relu(att1[b,p,a] + att2[b,a]) * fw[a] + fb ; softmax over p
__global__ __launch_bounds__(256)
void attn_kernel(const bf16* __restrict__ att1, const float* __restrict__ att2,
                 const float* __restrict__ fullW, const float* __restrict__ fullB,
                 float* __restrict__ alpha, float* __restrict__ alphas_out,
                 const int* __restrict__ lens, int tcur)
{
    __shared__ float a2[AAU];
    __shared__ float fw[AAU];
    __shared__ float red[256];
    const int b = blockIdx.x;
    const int tid = threadIdx.x;
    for (int i = tid; i < AAU; i += 256) { a2[i] = att2[b * AAU + i]; fw[i] = fullW[i]; }
    __syncthreads();

    float myv = 0.f;
    const int p = tid;
    if (p < PP) {
        const bf16* row = att1 + ((size_t)b * PP + p) * AAU;
        float s = 0.f;
        for (int a = 0; a < AAU; a += 8) {
            v8bf v = *(const v8bf*)(row + a);
#pragma unroll
            for (int j = 0; j < 8; ++j) {
                float t = (float)v[j] + a2[a + j];
                s += fmaxf(t, 0.f) * fw[a + j];
            }
        }
        myv = s + fullB[0];
    }
    red[tid] = (p < PP) ? myv : -3.0e38f;
    __syncthreads();
    for (int s = 128; s > 0; s >>= 1) {
        if (tid < s) red[tid] = fmaxf(red[tid], red[tid + s]);
        __syncthreads();
    }
    const float mx = red[0];
    __syncthreads();
    const float ex = (p < PP) ? __expf(myv - mx) : 0.f;
    red[tid] = ex;
    __syncthreads();
    for (int s = 128; s > 0; s >>= 1) {
        if (tid < s) red[tid] += red[tid + s];
        __syncthreads();
    }
    const float inv = 1.f / red[0];
    if (p < PP) {
        const float al = ex * inv;
        alpha[b * PP + p] = al;
        const float m = (tcur < lens[b] - 1) ? 1.f : 0.f;
        alphas_out[((size_t)b * TT + tcur) * PP + p] = al * m;
    }
}

// awe[b,c] = sum_p alpha[b,p] * enc[b,p,c]
__global__ __launch_bounds__(256)
void awe_kernel(const bf16* __restrict__ enc, const float* __restrict__ alpha,
                float* __restrict__ awe)
{
    __shared__ float al[PP];
    const int b = blockIdx.y;
    const int c = blockIdx.x * 256 + threadIdx.x;
    if (threadIdx.x < PP) al[threadIdx.x] = alpha[b * PP + threadIdx.x];
    __syncthreads();
    const bf16* base = enc + (size_t)b * PP * ENCC + c;
    float s = 0.f;
    for (int p = 0; p < PP; ++p) s += al[p] * (float)base[(size_t)p * ENCC];
    awe[(size_t)b * ENCC + c] = s;
}

// xh[b,:] = [ emb_W[cap[b,t]] (E) , gate*awe (ENC) , h (D) ]  as bf16
__global__ __launch_bounds__(256)
void build_xh(const float* __restrict__ embW, const int* __restrict__ caps,
              const float* __restrict__ gate, const float* __restrict__ awe,
              const float* __restrict__ h, bf16* __restrict__ xh, int tcur)
{
    const int b = blockIdx.x;
    const int cap = caps[b * TCAPP + tcur];
    for (int j = threadIdx.x; j < XHK; j += 256) {
        float v;
        if (j < EE) v = embW[(size_t)cap * EE + j];
        else if (j < EE + ENCC) { int jj = j - EE; v = gate[(size_t)b * ENCC + jj] * awe[(size_t)b * ENCC + jj]; }
        else v = h[(size_t)b * DD + (j - EE - ENCC)];
        xh[(size_t)b * XHK + j] = (bf16)v;
    }
}

// LSTM pointwise; ragged-batch mask keeps h,c frozen for finished rows
__global__ __launch_bounds__(256)
void lstm_kernel(const float* __restrict__ g, float* __restrict__ h,
                 float* __restrict__ c, bf16* __restrict__ hb,
                 const int* __restrict__ lens, int tcur)
{
    const int i = blockIdx.x * 256 + threadIdx.x;
    if (i >= BB * DD) return;
    const int b = i >> 9;
    const int j = i & (DD - 1);
    const size_t gb = (size_t)b * 4 * DD;
    const float ig = sigf(g[gb + j]);
    const float fg = sigf(g[gb + DD + j]);
    const float gg = tanhf(g[gb + 2 * DD + j]);
    const float og = sigf(g[gb + 3 * DD + j]);
    const float cn = fg * c[i] + ig * gg;
    const float hn = og * tanhf(cn);
    const bool act = tcur < lens[b] - 1;
    const float hv = act ? hn : h[i];
    const float cv = act ? cn : c[i];
    h[i] = hv; c[i] = cv; hb[i] = (bf16)hv;
}

// ================= host orchestration =================
static inline size_t aup(size_t x) { return (x + 255) & ~(size_t)255; }

extern "C" void kernel_launch(void* const* d_in, const int* in_sizes, int n_in,
                              void* d_out, int out_size, void* d_ws, size_t ws_size,
                              hipStream_t stream) {
    const float* encoder_out = (const float*)d_in[0];
    const float* box_feature = (const float*)d_in[1];
    const int*   caps        = (const int*)d_in[2];
    const int*   lens        = (const int*)d_in[3];
    const float* emb_W       = (const float*)d_in[4];
    const float* enc_att_W   = (const float*)d_in[5];
    const float* enc_att_b   = (const float*)d_in[6];
    const float* dec_att_W   = (const float*)d_in[7];
    const float* dec_att_b   = (const float*)d_in[8];
    const float* full_att_W  = (const float*)d_in[9];
    const float* full_att_b  = (const float*)d_in[10];
    const float* init_h_W    = (const float*)d_in[11];
    const float* init_h_b    = (const float*)d_in[12];
    const float* init_c_W    = (const float*)d_in[13];
    const float* init_c_b    = (const float*)d_in[14];
    const float* f_beta_W    = (const float*)d_in[15];
    const float* f_beta_b    = (const float*)d_in[16];
    const float* W_ih        = (const float*)d_in[17];
    const float* b_ih        = (const float*)d_in[18];
    const float* W_hh        = (const float*)d_in[19];
    const float* b_hh        = (const float*)d_in[20];
    const float* fc_W        = (const float*)d_in[21];
    const float* fc_b        = (const float*)d_in[22];

    float* preds_out  = (float*)d_out;                               // [B,T,V]
    float* alphas_out = (float*)d_out + (size_t)BB * TT * VV;        // [B,T,P]

    // ---- workspace carve-up ----
    char* w = (char*)d_ws; size_t off = 0;
    auto carve = [&](size_t bytes) { char* p = w + off; off = aup(off + bytes); return p; };
    bf16* enc_bf   = (bf16*)carve((size_t)BB * PP * ENCC * 2);   // 113 MB
    bf16* att1_bf  = (bf16*)carve((size_t)BB * PP * AAU * 2);    // 28 MB
    bf16* fcW_t    = (bf16*)carve((size_t)VV * DD * 2);          // 30.7 MB (L2-resident)
    bf16* Wcomb_t  = (bf16*)carve((size_t)(4 * DD) * XHK * 2);   // 12.6 MB
    bf16* encatt_t = (bf16*)carve((size_t)AAU * ENCC * 2);
    bf16* decatt_t = (bf16*)carve((size_t)AAU * DD * 2);
    bf16* fbeta_t  = (bf16*)carve((size_t)ENCC * DD * 2);
    bf16* initH_t  = (bf16*)carve((size_t)DD * ENCC * 2);
    bf16* initC_t  = (bf16*)carve((size_t)DD * ENCC * 2);
    bf16* mean_bf  = (bf16*)carve((size_t)BB * ENCC * 2);
    float* h       = (float*)carve((size_t)BB * DD * 4);
    float* c       = (float*)carve((size_t)BB * DD * 4);
    bf16*  h_bf    = (bf16*)carve((size_t)BB * DD * 2);
    float* att2    = (float*)carve((size_t)BB * AAU * 4);
    float* alpha   = (float*)carve((size_t)BB * PP * 4);
    float* awe     = (float*)carve((size_t)BB * ENCC * 4);
    float* gate    = (float*)carve((size_t)BB * ENCC * 4);
    bf16*  xh      = (bf16*)carve((size_t)BB * XHK * 2);
    float* g       = (float*)carve((size_t)BB * 4 * DD * 4);
    float* bcomb   = (float*)carve((size_t)4 * DD * 4);
    (void)ws_size;

    auto gblk = [](long long n) { return (unsigned)((n + 255) / 256); };
    auto blkOf = [](int M, int N, int MT, int NT) {
        return (unsigned)((((M >> 4) / MT) * ((N >> 4) / NT) + 7) / 8);
    };

    // ---- prep (once per call) ----
    {
        long long tot = (long long)BB * PP * ENCC;
        build_enc<<<gblk(tot), 256, 0, stream>>>(encoder_out, box_feature, enc_bf, tot);
    }
    tconv<<<gblk((long long)DD * VV), 256, 0, stream>>>(fc_W, fcW_t, DD, VV, DD, 0);
    tconv<<<gblk((long long)(EE + ENCC) * (4 * DD)), 256, 0, stream>>>(W_ih, Wcomb_t, EE + ENCC, 4 * DD, XHK, 0);
    tconv<<<gblk((long long)DD * (4 * DD)), 256, 0, stream>>>(W_hh, Wcomb_t, DD, 4 * DD, XHK, EE + ENCC);
    tconv<<<gblk((long long)ENCC * AAU), 256, 0, stream>>>(enc_att_W, encatt_t, ENCC, AAU, ENCC, 0);
    tconv<<<gblk((long long)DD * AAU), 256, 0, stream>>>(dec_att_W, decatt_t, DD, AAU, DD, 0);
    tconv<<<gblk((long long)DD * ENCC), 256, 0, stream>>>(f_beta_W, fbeta_t, DD, ENCC, DD, 0);
    tconv<<<gblk((long long)ENCC * DD), 256, 0, stream>>>(init_h_W, initH_t, ENCC, DD, ENCC, 0);
    tconv<<<gblk((long long)ENCC * DD), 256, 0, stream>>>(init_c_W, initC_t, ENCC, DD, ENCC, 0);
    vadd<<<gblk(4 * DD), 256, 0, stream>>>(b_ih, b_hh, bcomb, 4 * DD);

    mean_kernel<<<dim3(ENCC / 256, BB), 256, 0, stream>>>(enc_bf, mean_bf);

    // h0 / c0 : [128,2048] @ [2048,512]
    gemm_bf16_blk<2,2><<<blkOf(BB, DD, 2, 2), 256, 0, stream>>>(mean_bf, ENCC, initH_t, ENCC,
        h, DD, 0, init_h_b, BB, DD, ENCC, 0, nullptr, 0);
    gemm_bf16_blk<2,2><<<blkOf(BB, DD, 2, 2), 256, 0, stream>>>(mean_bf, ENCC, initC_t, ENCC,
        c, DD, 0, init_c_b, BB, DD, ENCC, 0, nullptr, 0);
    f2b<<<gblk(BB * DD), 256, 0, stream>>>(h, h_bf, BB * DD);

    // att1 = enc @ enc_att_W + b   (loop-invariant; [27648,2048]@[2048,512] -> bf16)
    gemm_bf16_blk<4,4><<<blkOf(BB * PP, AAU, 4, 4), 256, 0, stream>>>(enc_bf, ENCC, encatt_t, ENCC,
        att1_bf, AAU, 1, enc_att_b, BB * PP, AAU, ENCC, 0, nullptr, 0);

    // ---- sequential decode loop ----
    for (int t = 0; t < TT; ++t) {
        // att2 = h @ dec_att_W + b
        gemm_bf16_blk<2,2><<<blkOf(BB, AAU, 2, 2), 256, 0, stream>>>(h_bf, DD, decatt_t, DD,
            att2, AAU, 0, dec_att_b, BB, AAU, DD, 0, nullptr, 0);
        // e / softmax / alpha (+ masked alphas output)
        attn_kernel<<<BB, 256, 0, stream>>>(att1_bf, att2, full_att_W, full_att_b,
            alpha, alphas_out, lens, t);
        // awe = alpha . enc
        awe_kernel<<<dim3(ENCC / 256, BB), 256, 0, stream>>>(enc_bf, alpha, awe);
        // gate = sigmoid(h @ f_beta_W + b)
        gemm_bf16_blk<2,2><<<blkOf(BB, ENCC, 2, 2), 256, 0, stream>>>(h_bf, DD, fbeta_t, DD,
            gate, ENCC, 0, f_beta_b, BB, ENCC, DD, 1, nullptr, 0);
        // xh = [emb_t, gate*awe, h]
        build_xh<<<BB, 256, 0, stream>>>(emb_W, caps, gate, awe, h, xh, t);
        // g = xh @ [W_ih;W_hh] + (b_ih + b_hh)
        gemm_bf16_blk<2,2><<<blkOf(BB, 4 * DD, 2, 2), 256, 0, stream>>>(xh, XHK, Wcomb_t, XHK,
            g, 4 * DD, 0, bcomb, BB, 4 * DD, XHK, 0, nullptr, 0);
        // LSTM pointwise + ragged mask; refresh h_bf
        lstm_kernel<<<gblk(BB * DD), 256, 0, stream>>>(g, h, c, h_bf, lens, t);
        // preds[:,t,:] = mask * (h @ fc_W + fc_b)  -- N/16=1875=3*5^4, so NT=3
        gemm_bf16_blk<4,3><<<blkOf(BB, VV, 4, 3), 256, 0, stream>>>(h_bf, DD, fcW_t, DD,
            preds_out + (size_t)t * VV, (long long)TT * VV, 0, fc_b,
            BB, VV, DD, 0, lens, t);
    }
    (void)in_sizes; (void)n_in; (void)out_size;
}
